// GatNet_48524540510798
// MI455X (gfx1250) — compile-verified, matrix operand order checked by reading, not run on hardware
//
#include <hip/hip_runtime.h>
#include <hip/hip_bf16.h>

typedef __bf16 v16bf __attribute__((ext_vector_type(16)));
typedef __bf16 v4bf  __attribute__((ext_vector_type(4)));
typedef float  v8f   __attribute__((ext_vector_type(8)));

#define NEG_SLOPE 0.2f

// ---------------------------------------------------------------------------
// Fill kernel (needed for -inf init; also used for zeros to stay uniform)
// ---------------------------------------------------------------------------
__global__ void fill_kernel(float* __restrict__ p, float v, long long n) {
    long long i = (long long)blockIdx.x * blockDim.x + threadIdx.x;
    if (i < n) p[i] = v;
}

// ---------------------------------------------------------------------------
// WMMA GEMM:  Hout[M,64] = X[M,KT] @ W[KT,64]   (fp32 in, bf16 WMMA, f32 acc)
// Block = 256 threads (8 wave32s). Block tile: 128 rows x 64 cols.
// Wave w owns rows [w*16, w*16+16) -> 4 16x16 f32 accumulators.
// W preloaded (transposed, bf16) to LDS once; A tiles double-buffered,
// staged as float4 global loads -> packed bf16x4 LDS stores.
// ---------------------------------------------------------------------------
template <int KT>
__global__ __launch_bounds__(256) void gemm_wmma_kernel(
    const float* __restrict__ X, const float* __restrict__ W,
    float* __restrict__ Hout, int M) {
    __shared__ __bf16 wlds[64 * KT];       // [n][k]  (transposed W)
    __shared__ __bf16 alds[2][128 * 32];   // double-buffered A tile [m][k]

    const int tid  = threadIdx.x;
    const int lane = tid & 31;
    const int wave = tid >> 5;
    const int mblk = blockIdx.x * 128;

    // Coalesced W preload: linear global reads, transposed LDS scatter.
    for (int i = tid; i < KT * 64; i += 256) {
        int k = i >> 6, n = i & 63;
        wlds[n * KT + k] = (__bf16)W[i];
    }

    // A staging geometry: thread covers (row r0 + rr, cols c0..c0+3)
    const int r0 = tid >> 3;          // 0..31
    const int c0 = (tid & 7) << 2;    // 0,4,...,28

    auto stage = [&](int k0, int buf) {
#pragma unroll
        for (int rr = 0; rr < 128; rr += 32) {
            int gm = mblk + r0 + rr;
            gm = gm < M ? gm : M - 1;   // clamp: rows >= M never stored
            const float4 f =
                *(const float4*)(X + (size_t)gm * KT + k0 + c0);
            v4bf b;
            b[0] = (__bf16)f.x; b[1] = (__bf16)f.y;
            b[2] = (__bf16)f.z; b[3] = (__bf16)f.w;
            *(v4bf*)(&alds[buf][(r0 + rr) * 32 + c0]) = b;
        }
    };

    v8f acc[4];
#pragma unroll
    for (int t = 0; t < 4; ++t)
#pragma unroll
        for (int v = 0; v < 8; ++v) acc[t][v] = 0.0f;

    const int row16 = lane & 15;          // row in strip / output col
    const int kb    = (lane >> 4) * 8;    // K sub-base per ISA bf16 layout

    stage(0, 0);
    int p = 0;
    for (int k0 = 0; k0 < KT; k0 += 32, p ^= 1) {
        __syncthreads();   // staging of buffer p complete; p^1 free
        if (k0 + 32 < KT) stage(k0 + 32, p ^ 1);

        // A fragment: lane holds row (wave*16 + row16); K pattern:
        // vgpr0..3 -> K = kb+{0..7}; vgpr4..7 -> K = 16+kb+{0..7}
        const __bf16* ab = &alds[p][(wave * 16 + row16) * 32];
        v16bf afrag;
#pragma unroll
        for (int v = 0; v < 4; ++v) {
            afrag[2 * v]         = ab[kb + 2 * v];
            afrag[2 * v + 1]     = ab[kb + 2 * v + 1];
            afrag[8 + 2 * v]     = ab[16 + kb + 2 * v];
            afrag[8 + 2 * v + 1] = ab[16 + kb + 2 * v + 1];
        }

#pragma unroll
        for (int nt = 0; nt < 4; ++nt) {
            const __bf16* bb = &wlds[(nt * 16 + row16) * KT + k0];
            v16bf bfrag;
#pragma unroll
            for (int v = 0; v < 4; ++v) {
                bfrag[2 * v]         = bb[kb + 2 * v];
                bfrag[2 * v + 1]     = bb[kb + 2 * v + 1];
                bfrag[8 + 2 * v]     = bb[16 + kb + 2 * v];
                bfrag[8 + 2 * v + 1] = bb[16 + kb + 2 * v + 1];
            }
            acc[nt] = __builtin_amdgcn_wmma_f32_16x16x32_bf16(
                false, afrag, false, bfrag, (short)0, acc[nt], false, false);
        }
    }

    // C/D layout: vgpr v, lanes 0-15 -> M=v, lanes 16-31 -> M=8+v; N=lane&15
    const int rbase = mblk + wave * 16 + (lane >> 4) * 8;
#pragma unroll
    for (int nt = 0; nt < 4; ++nt)
#pragma unroll
        for (int v = 0; v < 8; ++v) {
            int r = rbase + v;
            if (r < M) Hout[(size_t)r * 64 + nt * 16 + row16] = acc[nt][v];
        }
}

// ---------------------------------------------------------------------------
// Per-node attention scores: s[n,h] = <h[n,h,:], a_src[h,:]>, same for d.
// ---------------------------------------------------------------------------
template <int H, int C>
__global__ void scores_kernel(const float* __restrict__ h,
                              const float* __restrict__ a_src,
                              const float* __restrict__ a_dst,
                              float* __restrict__ s, float* __restrict__ d,
                              int Nn) {
    int n = blockIdx.x * blockDim.x + threadIdx.x;
    if (n >= Nn) return;
    const float4* hp = (const float4*)(h + (size_t)n * 64);
#pragma unroll
    for (int hh = 0; hh < H; ++hh) {
        float ss = 0.f, dd = 0.f;
#pragma unroll
        for (int c4 = 0; c4 < C / 4; ++c4) {
            float4 v = hp[(hh * C) / 4 + c4];
            const float* as = a_src + hh * C + c4 * 4;
            const float* ad = a_dst + hh * C + c4 * 4;
            ss += v.x * as[0] + v.y * as[1] + v.z * as[2] + v.w * as[3];
            dd += v.x * ad[0] + v.y * ad[1] + v.z * ad[2] + v.w * ad[3];
        }
        s[(size_t)n * H + hh] = ss;
        d[(size_t)n * H + hh] = dd;
    }
}

__device__ __forceinline__ void edge_endpoints(const int* __restrict__ ei,
                                               long long e, long long E,
                                               int& src, int& dst) {
    if (e < E) { src = ei[e]; dst = ei[E + e]; }
    else       { src = (int)(e - E); dst = src; }   // self loops
}

// Pass 1: m[dst,h] = max over incoming edges of leakyrelu(s[src]+d[dst])
template <int H>
__global__ void edge_max_kernel(const int* __restrict__ ei,
                                const float* __restrict__ s,
                                const float* __restrict__ d,
                                float* __restrict__ m,
                                long long E, long long Etot) {
    long long i = (long long)blockIdx.x * blockDim.x + threadIdx.x;
    if (i >= Etot * H) return;
    long long e = i / H; int hh = (int)(i % H);
    int src, dst; edge_endpoints(ei, e, E, src, dst);
    float v = s[(size_t)src * H + hh] + d[(size_t)dst * H + hh];
    v = v > 0.f ? v : NEG_SLOPE * v;
    atomicMax(&m[(size_t)dst * H + hh], v);
}

// Pass 2: z[dst,h] = sum exp(e - m[dst,h])
template <int H>
__global__ void edge_sum_kernel(const int* __restrict__ ei,
                                const float* __restrict__ s,
                                const float* __restrict__ d,
                                const float* __restrict__ m,
                                float* __restrict__ z,
                                long long E, long long Etot) {
    long long i = (long long)blockIdx.x * blockDim.x + threadIdx.x;
    if (i >= Etot * H) return;
    long long e = i / H; int hh = (int)(i % H);
    int src, dst; edge_endpoints(ei, e, E, src, dst);
    float v = s[(size_t)src * H + hh] + d[(size_t)dst * H + hh];
    v = v > 0.f ? v : NEG_SLOPE * v;
    atomicAdd(&z[(size_t)dst * H + hh], __expf(v - m[(size_t)dst * H + hh]));
}

// Pass 3: out[dst,h,:] += alpha * hfeat[src,h,:]   (float4 gathers)
template <int H, int C>
__global__ void edge_agg_kernel(const int* __restrict__ ei,
                                const float* __restrict__ s,
                                const float* __restrict__ d,
                                const float* __restrict__ m,
                                const float* __restrict__ z,
                                const float* __restrict__ hfeat,
                                float* __restrict__ out,
                                long long E, long long Etot) {
    long long i = (long long)blockIdx.x * blockDim.x + threadIdx.x;
    if (i >= Etot * H) return;
    long long e = i / H; int hh = (int)(i % H);
    int src, dst; edge_endpoints(ei, e, E, src, dst);
    float v = s[(size_t)src * H + hh] + d[(size_t)dst * H + hh];
    v = v > 0.f ? v : NEG_SLOPE * v;
    float alpha = __expf(v - m[(size_t)dst * H + hh]) / z[(size_t)dst * H + hh];
    const float4* hs = (const float4*)(hfeat + (size_t)src * 64 + hh * C);
    float*        o  = out + (size_t)dst * 64 + hh * C;
#pragma unroll
    for (int c4 = 0; c4 < C / 4; ++c4) {
        float4 hv = hs[c4];
        atomicAdd(&o[4 * c4 + 0], alpha * hv.x);
        atomicAdd(&o[4 * c4 + 1], alpha * hv.y);
        atomicAdd(&o[4 * c4 + 2], alpha * hv.z);
        atomicAdd(&o[4 * c4 + 3], alpha * hv.w);
    }
}

__global__ void elu_kernel(float* __restrict__ x, long long n) {
    long long i = (long long)blockIdx.x * blockDim.x + threadIdx.x;
    if (i >= n) return;
    float v = x[i];
    x[i] = v > 0.f ? v : (__expf(v) - 1.0f);
}

__global__ void logsoftmax_kernel(float* __restrict__ out, int Nn) {
    int n = blockIdx.x * blockDim.x + threadIdx.x;
    if (n >= Nn) return;
    float4* row = (float4*)(out + (size_t)n * 64);
    float4 r[16];
    float mx = -1e30f;
#pragma unroll
    for (int c = 0; c < 16; ++c) {
        r[c] = row[c];
        mx = fmaxf(mx, fmaxf(fmaxf(r[c].x, r[c].y), fmaxf(r[c].z, r[c].w)));
    }
    float sum = 0.f;
#pragma unroll
    for (int c = 0; c < 16; ++c)
        sum += __expf(r[c].x - mx) + __expf(r[c].y - mx) +
               __expf(r[c].z - mx) + __expf(r[c].w - mx);
    float lse = mx + __logf(sum);
#pragma unroll
    for (int c = 0; c < 16; ++c) {
        r[c].x -= lse; r[c].y -= lse; r[c].z -= lse; r[c].w -= lse;
        row[c] = r[c];
    }
}

// ---------------------------------------------------------------------------
extern "C" void kernel_launch(void* const* d_in, const int* in_sizes, int n_in,
                              void* d_out, int out_size, void* d_ws, size_t ws_size,
                              hipStream_t stream) {
    (void)n_in; (void)out_size; (void)ws_size;
    const float* nodes    = (const float*)d_in[0];
    const int*   ei       = (const int*)d_in[1];
    const float* W1       = (const float*)d_in[2];
    const float* att_src1 = (const float*)d_in[3];
    const float* att_dst1 = (const float*)d_in[4];
    const float* W2       = (const float*)d_in[5];
    const float* att_src2 = (const float*)d_in[6];
    const float* att_dst2 = (const float*)d_in[7];
    float* outp = (float*)d_out;

    const int       Nn   = in_sizes[0] / 256;       // 100000
    const long long E    = in_sizes[1] / 2;         // 3.2M
    const long long Etot = E + Nn;                  // + self loops

    // Workspace carve-up (fp32)
    float* h1 = (float*)d_ws;                 // N*64
    float* s1 = h1 + (size_t)Nn * 64;         // N*8
    float* d1 = s1 + (size_t)Nn * 8;
    float* m1 = d1 + (size_t)Nn * 8;
    float* z1 = m1 + (size_t)Nn * 8;
    float* x1 = z1 + (size_t)Nn * 8;          // N*64
    float* h2 = x1 + (size_t)Nn * 64;         // N*64
    float* s2 = h2 + (size_t)Nn * 64;         // N
    float* d2 = s2 + (size_t)Nn;
    float* m2 = d2 + (size_t)Nn;
    float* z2 = m2 + (size_t)Nn;

    const int BT = 256;
    auto blocks = [&](long long n) { return (unsigned)((n + BT - 1) / BT); };

    // ---- init accumulators ----
    fill_kernel<<<blocks((long long)Nn * 64), BT, 0, stream>>>(x1, 0.0f, (long long)Nn * 64);
    fill_kernel<<<blocks((long long)Nn * 64), BT, 0, stream>>>(outp, 0.0f, (long long)Nn * 64);
    fill_kernel<<<blocks((long long)Nn * 8), BT, 0, stream>>>(m1, -1e30f, (long long)Nn * 8);
    fill_kernel<<<blocks((long long)Nn * 8), BT, 0, stream>>>(z1, 0.0f, (long long)Nn * 8);
    fill_kernel<<<blocks(Nn), BT, 0, stream>>>(m2, -1e30f, Nn);
    fill_kernel<<<blocks(Nn), BT, 0, stream>>>(z2, 0.0f, Nn);

    const unsigned gemm_blocks = (unsigned)((Nn + 127) / 128);

    // ---- layer 1 ----
    gemm_wmma_kernel<256><<<gemm_blocks, 256, 0, stream>>>(nodes, W1, h1, Nn);
    scores_kernel<8, 8><<<blocks(Nn), BT, 0, stream>>>(h1, att_src1, att_dst1, s1, d1, Nn);
    edge_max_kernel<8><<<blocks(Etot * 8), BT, 0, stream>>>(ei, s1, d1, m1, E, Etot);
    edge_sum_kernel<8><<<blocks(Etot * 8), BT, 0, stream>>>(ei, s1, d1, m1, z1, E, Etot);
    edge_agg_kernel<8, 8><<<blocks(Etot * 8), BT, 0, stream>>>(ei, s1, d1, m1, z1, h1, x1, E, Etot);
    elu_kernel<<<blocks((long long)Nn * 64), BT, 0, stream>>>(x1, (long long)Nn * 64);

    // ---- layer 2 ----
    gemm_wmma_kernel<64><<<gemm_blocks, 256, 0, stream>>>(x1, W2, h2, Nn);
    scores_kernel<1, 64><<<blocks(Nn), BT, 0, stream>>>(h2, att_src2, att_dst2, s2, d2, Nn);
    edge_max_kernel<1><<<blocks(Etot), BT, 0, stream>>>(ei, s2, d2, m2, E, Etot);
    edge_sum_kernel<1><<<blocks(Etot), BT, 0, stream>>>(ei, s2, d2, m2, z2, E, Etot);
    edge_agg_kernel<1, 64><<<blocks(Etot), BT, 0, stream>>>(ei, s2, d2, m2, z2, h2, outp, E, Etot);

    // ---- log softmax (in place on d_out) ----
    logsoftmax_kernel<<<blocks(Nn), BT, 0, stream>>>(outp, Nn);
}
